// PreciseDTI_47682726920334
// MI455X (gfx1250) — matrix operand order repaired and checked by reading, not compiled, not run on hardware
//
#include <hip/hip_runtime.h>
#include <hip/hip_bf16.h>
#include <math.h>

// ---------------- CDNA5 WMMA types ----------------
typedef __attribute__((ext_vector_type(16))) __bf16   v16bf;
typedef __attribute__((ext_vector_type(8)))  float    v8f;
typedef __attribute__((ext_vector_type(8)))  unsigned v8u;

__device__ __forceinline__ unsigned short f2bf(float f) {
    unsigned u = __builtin_bit_cast(unsigned, f);
    return (unsigned short)((u + 0x7FFFu + ((u >> 16) & 1u)) >> 16);  // RNE
}
__device__ __forceinline__ unsigned pack_bf16(float a, float b) {
    return (unsigned)f2bf(a) | ((unsigned)f2bf(b) << 16);
}

// ---- gfx1250 async global->LDS copy (ASYNCcnt path), 16B per lane ----
__device__ __forceinline__ void async_cp16(unsigned lds_byte_off, const void* gptr) {
    unsigned long long ga = (unsigned long long)gptr;
    asm volatile("global_load_async_to_lds_b128 %0, %1, off"
                 :: "v"(lds_byte_off), "v"(ga) : "memory");
}
__device__ __forceinline__ unsigned lds_off(const void* p) {
    return (unsigned)(unsigned long long)p;   // flat addr [31:0] == LDS offset
}
template <int N>
__device__ __forceinline__ void wait_asynccnt() {
    asm volatile("s_wait_asynccnt %0" :: "i"(N) : "memory");
}

// ---------------- pre-conversion kernels ----------------
__global__ void dti_conv_pairs(const float* __restrict__ in, unsigned* __restrict__ out, int nPairs) {
    int i = blockIdx.x * blockDim.x + threadIdx.x;
    if (i < nPairs) out[i] = pack_bf16(in[2 * i], in[2 * i + 1]);
}
// Wt[n][k] = bf16(W[k][n]), 384x384
__global__ void dti_conv_wt(const float* __restrict__ W, unsigned short* __restrict__ Wt) {
    int i = blockIdx.x * blockDim.x + threadIdx.x;   // 147456
    int n = i / 384, k = i % 384;
    Wt[(size_t)n * 384 + k] = f2bf(W[(size_t)k * 384 + n]);
}

// =====================================================================
// GEMM C = act(A[bf16, Mx384] @ Wt[bf16, 384x384 pre-transposed])
// 64x64 tile, 8 waves (4x2), wave = 16x32 (two 16x16 frags).
// Double-buffered LDS filled by global_load_async_to_lds_b128:
//   1 async op per thread per operand per tile -> s_wait_asynccnt 2 pipelines
//   the next tile's copy under the current tile's WMMAs.
// store mode 0: bf16 row-major (ld 384); mode 1: bf16 transposed (ld 16384)
// =====================================================================
__global__ __launch_bounds__(256)
void dti_gemm384_wmma(const unsigned short* __restrict__ A,
                      const unsigned short* __restrict__ Wt,
                      unsigned short* __restrict__ C, int act, int tmode)
{
    constexpr int K = 384;
    __shared__ unsigned ldsA[2][1024];   // 64 rows x 32 k (bf16) = 4KB / buf
    __shared__ unsigned ldsB[2][1024];   // 64 cols x 32 k (bf16), n-major
    const int tid   = threadIdx.x;
    const int mBase = (blockIdx.x / 6) * 64, nBase = (blockIdx.x % 6) * 64;
    const int lane = tid & 31, wave = tid >> 5;
    const int wm = wave >> 1,  wn = wave & 1;
    const int hi = lane >> 4,  lr = lane & 15;
    const int rr = tid >> 2, seg = tid & 3;          // 64 rows x 4 x 16B segments

    const unsigned a0 = lds_off(&ldsA[0][0]), a1 = lds_off(&ldsA[1][0]);
    const unsigned b0 = lds_off(&ldsB[0][0]), b1 = lds_off(&ldsB[1][0]);

    auto issue = [&](int bi, int kt) {
        async_cp16((bi ? a1 : a0) + rr * 64 + seg * 16,
                   (const char*)A  + ((size_t)(mBase + rr) * K + kt) * 2 + seg * 16);
        async_cp16((bi ? b1 : b0) + rr * 64 + seg * 16,
                   (const char*)Wt + ((size_t)(nBase + rr) * K + kt) * 2 + seg * 16);
    };

    v8f acc[2] = {};
    issue(0, 0);
    int bi = 0;
    for (int kt = 0; kt < K; kt += 32, bi ^= 1) {
        if (kt + 32 < K) { issue(bi ^ 1, kt + 32); wait_asynccnt<2>(); }
        else             { wait_asynccnt<0>(); }
        __syncthreads();                              // all waves' copies visible

        v8u au;
        #pragma unroll
        for (int v = 0; v < 8; ++v)
            au[v] = ldsA[bi][(wm * 16 + lr) * 16 + v + (v >= 4 ? 4 : 0) + 4 * hi];
        v16bf afrag = __builtin_bit_cast(v16bf, au);
        #pragma unroll
        for (int f = 0; f < 2; ++f) {
            v8u bu;
            #pragma unroll
            for (int v = 0; v < 8; ++v)
                bu[v] = ldsB[bi][(wn * 32 + f * 16 + lr) * 16 + v + (v >= 4 ? 4 : 0) + 4 * hi];
            v16bf bfrag = __builtin_bit_cast(v16bf, bu);
            acc[f] = __builtin_amdgcn_wmma_f32_16x16x32_bf16(
                false, afrag, false, bfrag, (short)0, acc[f], false, false);
        }
        __syncthreads();                              // done reading before reuse
    }

    #pragma unroll
    for (int f = 0; f < 2; ++f)
        #pragma unroll
        for (int v = 0; v < 8; ++v) {
            int row = mBase + wm * 16 + v + 8 * hi;
            int col = nBase + wn * 32 + f * 16 + lr;
            float x = acc[f][v];
            if (act) x = 0.5f * x * (1.0f + erff(x * 0.70710678118654752f));
            if (tmode) C[(size_t)col * 16384 + row] = f2bf(x);
            else       C[(size_t)row * 384   + col] = f2bf(x);
        }
}

// =====================================================================
// Patch pooling: S[b,512,384] = inball[b,512,4096] @ E[b,4096,384]
// Mask tile synthesized in LDS; Et (bf16, dim-major 384x16384) tile is a
// contiguous async copy, double-buffered. 4 x 8 x 6 = 192 workgroups.
// =====================================================================
__global__ __launch_bounds__(256)
void dti_patch_pool_wmma(const float* __restrict__ pos,
                         const unsigned short* __restrict__ Et,
                         float* __restrict__ S)
{
    __shared__ unsigned ldsA[1024];      // mask: 64 centers x 32 pts (bf16)
    __shared__ unsigned ldsB[2][1024];   // emb:  64 dims x 32 pts (bf16)
    __shared__ float cpos[64 * 3];
    __shared__ float ppos[32 * 3];

    const int tid = threadIdx.x;
    const int b = blockIdx.x / 48, rem = blockIdx.x % 48;
    const int p0 = (rem / 6) * 64, nBase = (rem % 6) * 64;
    const int lane = tid & 31, wave = tid >> 5;
    const int wm = wave >> 1,  wn = wave & 1;
    const int hi = lane >> 4,  lr = lane & 15;
    const int rr = tid >> 2, seg = tid & 3;

    const unsigned bB0 = lds_off(&ldsB[0][0]), bB1 = lds_off(&ldsB[1][0]);
    auto issueB = [&](int bi, int kt) {
        async_cp16((bi ? bB1 : bB0) + rr * 64 + seg * 16,
                   (const char*)Et + ((size_t)(nBase + rr) * 16384 + b * 4096 + kt) * 2 + seg * 16);
    };

    if (tid < 192) {   // 64 centers * xyz; center p -> point p*8 (NB/P == 8)
        int c = tid / 3, d = tid % 3;
        cpos[tid] = pos[((size_t)(b * 4096 + (p0 + c) * 8)) * 3 + d];
    }

    v8f acc[2] = {};
    issueB(0, 0);
    int bi = 0;
    for (int kt = 0; kt < 4096; kt += 32, bi ^= 1) {
        bool nxt = (kt + 32 < 4096);
        if (nxt) issueB(bi ^ 1, kt + 32);
        if (tid < 96)
            ppos[tid] = pos[((size_t)(b * 4096 + kt + tid / 3)) * 3 + (tid % 3)];
        __syncthreads();                     // ppos visible; prev wmma done

        for (int e = tid; e < 1024; e += 256) {
            int c = e >> 4, kp = e & 15;
            float cx = cpos[c * 3], cy = cpos[c * 3 + 1], cz = cpos[c * 3 + 2];
            unsigned m = 0;
            #pragma unroll
            for (int j = 0; j < 2; ++j) {
                int p = 2 * kp + j;
                float dx = ppos[p * 3] - cx, dy = ppos[p * 3 + 1] - cy, dz = ppos[p * 3 + 2] - cz;
                if (dx * dx + dy * dy + dz * dz <= 49.0f)   // RADIUS^2; 1.0bf16=0x3F80
                    m |= (0x3F80u << (16 * j));
            }
            ldsA[e] = m;
        }
        if (nxt) wait_asynccnt<1>(); else wait_asynccnt<0>();
        __syncthreads();                     // mask + emb tile ready

        v8u au;
        #pragma unroll
        for (int v = 0; v < 8; ++v)
            au[v] = ldsA[(wm * 16 + lr) * 16 + v + (v >= 4 ? 4 : 0) + 4 * hi];
        v16bf afrag = __builtin_bit_cast(v16bf, au);
        #pragma unroll
        for (int f = 0; f < 2; ++f) {
            v8u bu;
            #pragma unroll
            for (int v = 0; v < 8; ++v)
                bu[v] = ldsB[bi][(wn * 32 + f * 16 + lr) * 16 + v + (v >= 4 ? 4 : 0) + 4 * hi];
            v16bf bfrag = __builtin_bit_cast(v16bf, bu);
            acc[f] = __builtin_amdgcn_wmma_f32_16x16x32_bf16(
                false, afrag, false, bfrag, (short)0, acc[f], false, false);
        }
        __syncthreads();
    }

    #pragma unroll
    for (int f = 0; f < 2; ++f)
        #pragma unroll
        for (int v = 0; v < 8; ++v) {
            int row = p0 + wm * 16 + v + 8 * hi;
            int col = nBase + wn * 32 + f * 16 + lr;
            S[((size_t)(b * 512 + row)) * 384 + col] = acc[f][v];
        }
}

// ---------------- per-center counts (for patch_mask) ----------------
__global__ __launch_bounds__(256)
void dti_counts(const float* __restrict__ pos, float* __restrict__ counts)
{
    int w = blockIdx.x * 8 + (threadIdx.x >> 5);
    int lane = threadIdx.x & 31;
    int b = w >> 9, p = w & 511;
    size_t ci = (size_t)(b * 4096 + p * 8) * 3;
    float cx = pos[ci], cy = pos[ci + 1], cz = pos[ci + 2];
    float cnt = 0.f;
    for (int i = lane; i < 4096; i += 32) {
        size_t pi = (size_t)(b * 4096 + i) * 3;
        float dx = pos[pi] - cx, dy = pos[pi + 1] - cy, dz = pos[pi + 2] - cz;
        cnt += (dx * dx + dy * dy + dz * dz <= 49.0f) ? 1.0f : 0.0f;
    }
    for (int o = 16; o; o >>= 1) cnt += __shfl_xor(cnt, o, 32);
    if (lane == 0) counts[w] = cnt;
}

// ---------------- tiny MLPs ----------------
__global__ void dti_mlp_hidden(const float* __restrict__ code, const float* __restrict__ Wc1,
                               const float* __restrict__ bc1,  const float* __restrict__ drug,
                               const float* __restrict__ Wd1,  const float* __restrict__ bd1,
                               float* __restrict__ hq, float* __restrict__ hd)
{
    int idx = blockIdx.x * blockDim.x + threadIdx.x;
    if (idx < 1536) {
        int b = idx / 384, j = idx % 384;
        float acc = bc1[j];
        for (int k = 0; k < 768; ++k) acc += code[b * 768 + k] * Wc1[(size_t)k * 384 + j];
        hq[idx] = acc / (1.0f + expf(-acc));
    } else if (idx < 3072) {
        int i2 = idx - 1536, b = i2 / 384, j = i2 % 384;
        float acc = bd1[j];
        for (int k = 0; k < 2048; ++k) acc += drug[b * 2048 + k] * Wd1[(size_t)k * 384 + j];
        hd[i2] = acc / (1.0f + expf(-acc));
    }
}

__global__ void dti_mlp_out(const float* __restrict__ hq, const float* __restrict__ Wc2,
                            const float* __restrict__ bc2, const float* __restrict__ hd,
                            const float* __restrict__ Wd2, const float* __restrict__ bd2,
                            float* __restrict__ qdu, float* __restrict__ du)
{
    int idx = blockIdx.x * blockDim.x + threadIdx.x;
    if (idx < 1536) {
        int b = idx / 384, j = idx % 384;
        float acc = bc2[j];
        for (int k = 0; k < 384; ++k) acc += hq[b * 384 + k] * Wc2[(size_t)k * 384 + j];
        qdu[idx] = acc;
    } else if (idx < 3072) {
        int i2 = idx - 1536, b = i2 / 384, j = i2 % 384;
        float acc = bd2[j];
        for (int k = 0; k < 384; ++k) acc += hd[b * 384 + k] * Wd2[(size_t)k * 384 + j];
        du[i2] = acc;
    }
}

// ---------------- finalize ----------------
// normalize(S/max(c,1)) == normalize(S): counts only feed patch_mask.
__global__ __launch_bounds__(256)
void dti_finalize(const float* __restrict__ S, const float* __restrict__ qdu,
                  const float* __restrict__ du, const float* __restrict__ counts,
                  float* __restrict__ out)
{
    int w = blockIdx.x * 8 + (threadIdx.x >> 5);
    int lane = threadIdx.x & 31;
    int b = w >> 9;
    const float* Sp = S + (size_t)w * 384;
    const float* qp = qdu + b * 384;
    const float* dp = du + b * 384;
    float ss = 0, qq = 0, dd = 0, sq = 0, sd = 0;
    for (int i = lane; i < 384; i += 32) {
        float s = Sp[i], q = qp[i], dv = dp[i];
        ss += s * s; qq += q * q; dd += dv * dv; sq += s * q; sd += s * dv;
    }
    for (int o = 16; o; o >>= 1) {
        ss += __shfl_xor(ss, o, 32); qq += __shfl_xor(qq, o, 32); dd += __shfl_xor(dd, o, 32);
        sq += __shfl_xor(sq, o, 32); sd += __shfl_xor(sd, o, 32);
    }
    if (lane == 0) {
        float ns = fmaxf(sqrtf(ss), 1e-8f);
        float nq = fmaxf(sqrtf(qq), 1e-8f);
        float nd = fmaxf(sqrtf(dd), 1e-8f);
        out[w]        = sq / (ns * nq);
        out[2048 + w] = sd / (ns * nd);
        out[4096 + w] = counts[w] > 0.f ? 1.0f : 0.0f;
    }
}

extern "C" void kernel_launch(void* const* d_in, const int* in_sizes, int n_in,
                              void* d_out, int out_size, void* d_ws, size_t ws_size,
                              hipStream_t stream)
{
    const float* rec_h = (const float*)d_in[0];
    const float* pos   = (const float*)d_in[1];
    const float* drug  = (const float*)d_in[3];
    const float* code  = (const float*)d_in[4];
    const float* W1    = (const float*)d_in[5];
    const float* W2    = (const float*)d_in[6];
    const float* Wd1   = (const float*)d_in[7];
    const float* bd1   = (const float*)d_in[8];
    const float* Wd2   = (const float*)d_in[9];
    const float* bd2   = (const float*)d_in[10];
    const float* Wc1   = (const float*)d_in[11];
    const float* bc1   = (const float*)d_in[12];
    const float* Wc2   = (const float*)d_in[13];
    const float* bc2   = (const float*)d_in[14];
    float* out = (float*)d_out;

    char* w = (char*)d_ws;                                   // byte offsets
    unsigned short* Abf = (unsigned short*)(w);              // 16384*384 bf16 = 12,582,912 B
    unsigned short* W1t = (unsigned short*)(w + 12582912);   //   294,912 B
    unsigned short* W2t = (unsigned short*)(w + 12877824);   //   294,912 B
    unsigned short* H1  = (unsigned short*)(w + 13172736);   // 12,582,912 B
    unsigned short* Et  = (unsigned short*)(w + 25755648);   // 12,582,912 B (dim-major 384x16384)
    float* S      = (float*)(w + 38338560);                  // 3,145,728 B
    float* counts = (float*)(w + 41484288);                  //     8,192 B
    float* hq     = (float*)(w + 41492480);
    float* hd     = (float*)(w + 41498624);
    float* qdu    = (float*)(w + 41504768);
    float* du     = (float*)(w + 41510912);

    // one-time bf16 packing / weight transposes
    dti_conv_pairs<<<12288, 256, 0, stream>>>(rec_h, (unsigned*)Abf, 3145728);
    dti_conv_wt<<<576, 256, 0, stream>>>(W1, W1t);
    dti_conv_wt<<<576, 256, 0, stream>>>(W2, W2t);

    // rec_embed = gelu(rec_h @ W1) @ W2 ; Et stored dim-major for patch pool
    dti_gemm384_wmma<<<1536, 256, 0, stream>>>(Abf, W1t, H1, 1, 0);
    dti_gemm384_wmma<<<1536, 256, 0, stream>>>(H1,  W2t, Et, 0, 1);

    dti_patch_pool_wmma<<<192, 256, 0, stream>>>(pos, Et, S);
    dti_counts<<<256, 256, 0, stream>>>(pos, counts);

    dti_mlp_hidden<<<12, 256, 0, stream>>>(code, Wc1, bc1, drug, Wd1, bd1, hq, hd);
    dti_mlp_out<<<12, 256, 0, stream>>>(hq, Wc2, bc2, hd, Wd2, bd2, qdu, du);

    dti_finalize<<<256, 256, 0, stream>>>(S, qdu, du, counts, out);
}
